// RnnModel_12412455486143
// MI455X (gfx1250) — matrix operand order repaired and checked by reading, not compiled, not run on hardware
//
#include <hip/hip_runtime.h>
#include <hip/hip_bf16.h>

// ---------------- problem dims ----------------
#define T_STEPS 4096
#define F_IN    1024
#define H1      2048
#define H2      2048
#define G4      (4 * H2)   // 8192 gate rows
#define H3      1024
#define NBLK    64         // persistent blocks for the recurrence (H2/NBLK = 32 units/block)

typedef __attribute__((ext_vector_type(16))) _Float16 v16h;
typedef __attribute__((ext_vector_type(8)))  float    v8f;
typedef __attribute__((ext_vector_type(4)))  unsigned int tdm_u32x4;
typedef __attribute__((ext_vector_type(8)))  int          tdm_i32x8;
typedef __attribute__((ext_vector_type(4)))  int          tdm_i32x4;

// ---------------- workspace layout (bytes) ----------------
constexpr size_t OFF_XH   = 0;                                     // x   f16 [4096,1024]
constexpr size_t OFF_W1H  = OFF_XH   + (size_t)T_STEPS * F_IN * 2; // W1  f16 [2048,1024]
constexpr size_t OFF_ZH   = OFF_W1H  + (size_t)H1 * F_IN * 2;      // z   f16 [4096,2048]
constexpr size_t OFF_WIHH = OFF_ZH   + (size_t)T_STEPS * H1 * 2;   // Wih f16 [8192,2048]
constexpr size_t OFF_WHHH = OFF_WIHH + (size_t)G4 * H1 * 2;        // Whh f16 [8192,2048]
constexpr size_t OFF_G    = OFF_WHHH + (size_t)G4 * H2 * 2;        // G   f32 [4096,8192]
constexpr size_t OFF_H    = OFF_G    + (size_t)T_STEPS * G4 * 4;   // h double buffer f32 [2,2048]
constexpr size_t OFF_C    = OFF_H    + (size_t)2 * H2 * 4;         // c f32 [2048]
constexpr size_t OFF_BAR  = OFF_C    + (size_t)H2 * 4;             // barrier counter

// ---------------- fp32 -> fp16 convert ----------------
__global__ __launch_bounds__(256) void cvt_f32_to_f16(const float* __restrict__ src,
                                                      _Float16* __restrict__ dst, int n) {
  int i = blockIdx.x * blockDim.x + threadIdx.x;
  int stride = gridDim.x * blockDim.x;
  for (; i < n; i += stride) dst[i] = (_Float16)src[i];
}

// ---------------- WMMA fragment loaders (contiguous-k order, same for A and B) ----------------
// A 16x32 f16 from LDS: lanes 0-15 -> M=lane, K = k0..k0+15 contiguous;
// lanes 16-31 -> same M, K = k0+16..k0+31.  Emits ds_load_b128 pairs.
__device__ inline v16h load_a_frag_lds(const _Float16* p, int ld, int k0, int lane) {
  int m  = lane & 15;
  int kk = k0 + ((lane >> 4) << 4);
  const _Float16* q = p + m * ld + kk;
  union { uint4 u[2]; v16h v; } r;
  r.u[0] = *(const uint4*)(q);
  r.u[1] = *(const uint4*)(q + 8);
  return r.v;
}
// B 32x16 f16 built from row-major weight W[N,K] (B = W^T): lanes 0-15 -> N=lane,
// K = k0..k0+15 contiguous; lanes 16-31 -> N=lane-16, K = k0+16..k0+31.
__device__ inline v16h load_b_frag(const _Float16* __restrict__ p, int ld,
                                   int n_base, int k0, int lane) {
  int n  = n_base + (lane & 15);
  int kk = k0 + ((lane >> 4) << 4);
  const _Float16* q = p + (size_t)n * ld + kk;
  union { uint4 u[2]; v16h v; } r;
  r.u[0] = *(const uint4*)(q);
  r.u[1] = *(const uint4*)(q + 8);
  return r.v;
}

#if __has_builtin(__builtin_amdgcn_tensor_load_to_lds)
// TDM: DMA `rows` x K f16 row-major tile from global into LDS (D# per ISA 8.3/8.4).
// Descriptor values are wave-uniform; EXEC is ignored by TENSOR_LOAD_TO_LDS.
// amdgpu-toolchain (clang-23) arity: (u32x4 g0, i32x8 g1, i32x4 g2, i32x4 g3, i32x8 g4, i32 cpol)
__device__ inline void tdm_load_rows_to_lds(const _Float16* gsrc, _Float16* ldst,
                                            int rows, int K) {
  unsigned long long ga = (unsigned long long)gsrc;
  unsigned int       la = (unsigned int)(unsigned long long)ldst;  // low 32b = LDS offset
  tdm_u32x4 g0 = {
      1u,                                                   // count=1, user descriptor
      la,                                                   // lds_addr
      (unsigned int)ga,                                     // global_addr[31:0]
      ((unsigned int)(ga >> 32) & 0x01FFFFFFu) | 0x80000000u // global_addr[56:32] | type=2
  };
  tdm_i32x8 g1 = {
      (int)(1u << 16),                                      // data_size=1 (2 bytes)
      (int)(((unsigned)K & 0xFFFFu) << 16),                 // tensor_dim0[15:0]
      (int)(((unsigned)K >> 16) | ((unsigned)rows << 16)),  // tensor_dim0[31:16] | tensor_dim1[15:0]
      (int)(((unsigned)K & 0xFFFFu) << 16),                 // tensor_dim1 hi=0 | tile_dim0=K
      rows,                                                 // tile_dim1=rows, tile_dim2=0
      K,                                                    // tensor_dim0_stride[31:0]
      0, 0                                                  // stride hi, tensor_dim1_stride
  };
  tdm_i32x4 gz  = {0, 0, 0, 0};
  tdm_i32x8 gz8 = {0, 0, 0, 0, 0, 0, 0, 0};
  __builtin_amdgcn_tensor_load_to_lds(g0, g1, gz, gz, gz8, 0);
}
#endif

// ---------------- tiled GEMM: C = act(A @ W^T + bias) ----------------
// Block tile: 64 M-rows x 128 N-cols, 256 threads = 8 waves.
// A strip [64 x K] staged in LDS (TDM, one 8-row slice per wave); each wave owns
// one 16-col strip and reuses its B fragment across 4 M-tiles -> 4 WMMA per B load.
// ACT=1: sigmoid -> f16 out (Phase A). ACT=0: +bias2, f32 out (Phase B).
template <int ACT>
__global__ __launch_bounds__(256) void gemm_tile_kernel(const _Float16* __restrict__ A,
                                                        const _Float16* __restrict__ W,
                                                        const float* __restrict__ bias1,
                                                        const float* __restrict__ bias2,
                                                        float* __restrict__ Cf,
                                                        _Float16* __restrict__ Ch,
                                                        int M, int N, int K) {
  extern __shared__ _Float16 lds_a[];          // [64][K]
  const int tid  = threadIdx.x;
  const int wave = tid >> 5;
  const int lane = tid & 31;
  const int mt = blockIdx.x * 64;
  const int nt = blockIdx.y * 128 + wave * 16;

#if __has_builtin(__builtin_amdgcn_tensor_load_to_lds)
  // each wave DMAs its own 8-row slice of the A strip
  tdm_load_rows_to_lds(A + (size_t)(mt + wave * 8) * K, lds_a + (size_t)wave * 8 * K, 8, K);
  __builtin_amdgcn_s_wait_tensorcnt(0);
#else
  {
    const uint4* s = (const uint4*)(A + (size_t)mt * K);
    uint4* d = (uint4*)lds_a;
    for (int i = tid; i < 8 * K; i += 256) d[i] = s[i];   // 64*K halves / 8 per uint4
  }
#endif
  __syncthreads();

  v8f acc0 = {}, acc1 = {}, acc2 = {}, acc3 = {};
  for (int k0 = 0; k0 < K; k0 += 32) {
    if (k0 + 64 < K)
      __builtin_prefetch((const char*)(W + (size_t)(nt + (lane & 15)) * K + k0 + 64), 0, 1);
    v16h b  = load_b_frag(W, K, nt, k0, lane);
    v16h a0 = load_a_frag_lds(lds_a + 0 * 16 * K, K, k0, lane);
    v16h a1 = load_a_frag_lds(lds_a + 1 * 16 * K, K, k0, lane);
    v16h a2 = load_a_frag_lds(lds_a + 2 * 16 * K, K, k0, lane);
    v16h a3 = load_a_frag_lds(lds_a + 3 * 16 * K, K, k0, lane);
    acc0 = __builtin_amdgcn_wmma_f32_16x16x32_f16(false, a0, false, b, (short)0, acc0, false, false);
    acc1 = __builtin_amdgcn_wmma_f32_16x16x32_f16(false, a1, false, b, (short)0, acc1, false, false);
    acc2 = __builtin_amdgcn_wmma_f32_16x16x32_f16(false, a2, false, b, (short)0, acc2, false, false);
    acc3 = __builtin_amdgcn_wmma_f32_16x16x32_f16(false, a3, false, b, (short)0, acc3, false, false);
  }

  const int n = nt + (lane & 15);
  const float bv = ACT ? bias1[n] : (bias1[n] + bias2[n]);
  const int mrow = ((lane >> 4) << 3);
  v8f accs[4] = {acc0, acc1, acc2, acc3};
  #pragma unroll
  for (int mi = 0; mi < 4; ++mi) {
    const int mbase = mt + mi * 16 + mrow;
    #pragma unroll
    for (int v = 0; v < 8; ++v) {
      float g = accs[mi][v] + bv;
      if (ACT) {
        float s = 1.f / (1.f + __expf(-g));
        Ch[(size_t)(mbase + v) * N + n] = (_Float16)s;
      } else {
        Cf[(size_t)(mbase + v) * N + n] = g;
      }
    }
  }
}

// ---------------- state init ----------------
__global__ __launch_bounds__(256) void init_state(float* hbuf, float* cbuf, int* bar) {
  int i = blockIdx.x * blockDim.x + threadIdx.x;
  if (i < 2 * H2) hbuf[i] = 0.f;
  if (i < H2)     cbuf[i] = 0.f;
  if (i == 0)     *bar = 0;
}

// ---------------- Phase C: sequential LSTM, persistent grid-synced kernel ----
// 64 blocks x 256 threads. Block b owns hidden units [b*32, b*32+32): 128 gate
// rows. Whh (f16, 32 MB) stays L2-resident; per-step gates = G[t] + h @ Whh^T.
__global__ __launch_bounds__(256, 1) void lstm_kernel(const float* __restrict__ G,
                                                      const _Float16* __restrict__ Whh,
                                                      float* __restrict__ hbuf,
                                                      float* __restrict__ cbuf,
                                                      int* __restrict__ bar) {
  __shared__ float h_s[H2];
  __shared__ float red[256];
  __shared__ float gate[128];
  const int tid = threadIdx.x;
  const int j0 = blockIdx.x * (H2 / NBLK);      // first hidden unit of this block
  const int row_idx = tid >> 1;                 // 0..127  (4 gates x 32 units)
  const int khalf = tid & 1;                    // split K=2048 into two halves
  const int row = (row_idx >> 5) * H2 + j0 + (row_idx & 31);  // row in [8192,2048]
  const _Float16* wrow = Whh + (size_t)row * H2 + khalf * (H2 / 2);

  for (int t = 0; t < T_STEPS; ++t) {
    const float* hin = hbuf + (t & 1) * H2;
    float* hout = hbuf + ((t + 1) & 1) * H2;

    for (int k = tid; k < H2; k += 256) h_s[k] = hin[k];
    __syncthreads();

    if (t + 1 < T_STEPS)
      __builtin_prefetch((const char*)(G + (size_t)(t + 1) * G4 + row), 0, 0);

    float acc = 0.f;
    const float* hh = h_s + khalf * (H2 / 2);
    #pragma unroll 4
    for (int k = 0; k < H2 / 2; k += 8) {
      union { uint4 u; _Float16 hv[8]; } w;
      w.u = *(const uint4*)(wrow + k);
      acc += (float)w.hv[0] * hh[k + 0] + (float)w.hv[1] * hh[k + 1]
           + (float)w.hv[2] * hh[k + 2] + (float)w.hv[3] * hh[k + 3]
           + (float)w.hv[4] * hh[k + 4] + (float)w.hv[5] * hh[k + 5]
           + (float)w.hv[6] * hh[k + 6] + (float)w.hv[7] * hh[k + 7];
    }
    red[tid] = acc;
    __syncthreads();

    if (tid < 128) {
      int r = (tid >> 5) * H2 + j0 + (tid & 31);
      gate[tid] = red[2 * tid] + red[2 * tid + 1] + G[(size_t)t * G4 + r];
    }
    __syncthreads();

    if (tid < 32) {  // one thread per hidden unit: i,f,g,o -> c,h
      float iv = 1.f / (1.f + __expf(-gate[tid]));
      float fv = 1.f / (1.f + __expf(-gate[32 + tid]));
      float gv = tanhf(gate[64 + tid]);
      float ov = 1.f / (1.f + __expf(-gate[96 + tid]));
      float cn = fv * cbuf[j0 + tid] + iv * gv;
      cbuf[j0 + tid] = cn;
      hout[j0 + tid] = ov * tanhf(cn);
    }

    __threadfence();
    __syncthreads();
    if (tid == 0) {
      __hip_atomic_fetch_add(bar, 1, __ATOMIC_ACQ_REL, __HIP_MEMORY_SCOPE_AGENT);
      const int target = NBLK * (t + 1);
      while (__hip_atomic_load(bar, __ATOMIC_ACQUIRE, __HIP_MEMORY_SCOPE_AGENT) < target)
        __builtin_amdgcn_s_sleep(2);
    }
    __syncthreads();
  }
}

// ---------------- Phase D: out = relu(h @ W2^T + b2) @ W3^T + b3 ----------------
__global__ __launch_bounds__(256) void head_kernel(const float* __restrict__ h,
                                                   const float* __restrict__ W2,
                                                   const float* __restrict__ b2,
                                                   const float* __restrict__ W3,
                                                   const float* __restrict__ b3,
                                                   float* __restrict__ out) {
  __shared__ float y_s[H3];
  __shared__ float red[256];
  const int tid = threadIdx.x;
  for (int j = tid; j < H3; j += 256) {
    const float* wr = W2 + (size_t)j * H2;
    float acc = 0.f;
    for (int k = 0; k < H2; ++k) acc += wr[k] * h[k];
    float v = acc + b2[j];
    y_s[j] = v > 0.f ? v : 0.f;
  }
  __syncthreads();
  float p = 0.f;
  for (int j = tid; j < H3; j += 256) p += y_s[j] * W3[j];
  red[tid] = p;
  __syncthreads();
  for (int s = 128; s > 0; s >>= 1) {
    if (tid < s) red[tid] += red[tid + s];
    __syncthreads();
  }
  if (tid == 0) out[0] = red[0] + b3[0];
}

// ---------------- launch ----------------
extern "C" void kernel_launch(void* const* d_in, const int* in_sizes, int n_in,
                              void* d_out, int out_size, void* d_ws, size_t ws_size,
                              hipStream_t stream) {
  (void)in_sizes; (void)n_in; (void)out_size; (void)ws_size;
  const float* x   = (const float*)d_in[0];
  const float* W1  = (const float*)d_in[1];
  const float* b1  = (const float*)d_in[2];
  const float* Wih = (const float*)d_in[3];
  const float* Whh = (const float*)d_in[4];
  const float* bih = (const float*)d_in[5];
  const float* bhh = (const float*)d_in[6];
  const float* W2  = (const float*)d_in[7];
  const float* b2  = (const float*)d_in[8];
  const float* W3  = (const float*)d_in[9];
  const float* b3  = (const float*)d_in[10];
  float* out = (float*)d_out;

  char* ws = (char*)d_ws;
  _Float16* Xh   = (_Float16*)(ws + OFF_XH);
  _Float16* W1h  = (_Float16*)(ws + OFF_W1H);
  _Float16* Zh   = (_Float16*)(ws + OFF_ZH);
  _Float16* Wihh = (_Float16*)(ws + OFF_WIHH);
  _Float16* Whhh = (_Float16*)(ws + OFF_WHHH);
  float*    Gb   = (float*)(ws + OFF_G);
  float*    hbuf = (float*)(ws + OFF_H);
  float*    cbuf = (float*)(ws + OFF_C);
  int*      bar  = (int*)(ws + OFF_BAR);

  init_state<<<(2 * H2 + 255) / 256, 256, 0, stream>>>(hbuf, cbuf, bar);

  cvt_f32_to_f16<<<4096, 256, 0, stream>>>(x,   Xh,   T_STEPS * F_IN);
  cvt_f32_to_f16<<<4096, 256, 0, stream>>>(W1,  W1h,  H1 * F_IN);
  cvt_f32_to_f16<<<4096, 256, 0, stream>>>(Wih, Wihh, G4 * H1);
  cvt_f32_to_f16<<<4096, 256, 0, stream>>>(Whh, Whhh, G4 * H2);

  // Phase A: Z = sigmoid(X @ W1^T + b1)  [4096 x 2048], LDS strip 64x1024 f16 = 128 KB
  gemm_tile_kernel<1><<<dim3(T_STEPS / 64, H1 / 128), 256, 64 * F_IN * 2, stream>>>(
      Xh, W1h, b1, b1, nullptr, Zh, T_STEPS, H1, F_IN);

  // Phase B: G = Z @ Wih^T + (bih + bhh) [4096 x 8192], LDS strip 64x2048 f16 = 256 KB
  gemm_tile_kernel<0><<<dim3(T_STEPS / 64, G4 / 128), 256, 64 * H1 * 2, stream>>>(
      Zh, Wihh, bih, bhh, Gb, nullptr, T_STEPS, G4, H1);

  // Phase C: sequential LSTM recurrence (persistent, grid-synced)
  lstm_kernel<<<NBLK, 256, 0, stream>>>(Gb, Whhh, hbuf, cbuf, bar);

  // Phase D: head (final h lives in hbuf[0] since T is even)
  head_kernel<<<1, 256, 0, stream>>>(hbuf, W2, b2, W3, b3, out);
}
